// Block_59700045414559
// MI455X (gfx1250) — matrix operand order repaired and checked by reading, not compiled, not run on hardware
//
#include <hip/hip_runtime.h>

typedef __attribute__((ext_vector_type(16))) _Float16 v16h;
typedef __attribute__((ext_vector_type(8)))  float    v8f;
typedef __attribute__((ext_vector_type(2)))  float    v2f;
typedef _Float16 f16;

#define kB  4
#define kNI 128
#define kNO 256
#define kL  2048
#define kNS 192
#define kE  16
#define kTD 1536

__device__ __forceinline__ float mishf(float x) {
    float sp = (x > 20.f) ? x : log1pf(__expf(x));
    return x * tanhf(sp);
}

__device__ __forceinline__ v16h h16zero() {
    v16h z;
#pragma unroll
    for (int i = 0; i < 16; i++) z[i] = (f16)0.f;
    return z;
}

// ---------------------------------------------------------------------------
// Prep: convert conv weights f32 [Co][Ci][T] -> f16 [Co][T][Ci]
// ---------------------------------------------------------------------------
__global__ void k_prep_convw(const float* __restrict__ w, f16* __restrict__ o,
                             int Co, int Ci, int T) {
    int i = blockIdx.x * blockDim.x + threadIdx.x;
    int n = Co * Ci * T;
    if (i >= n) return;
    int co = i / (Ci * T);
    int r  = i % (Ci * T);
    int ci = r / T;
    int t  = r % T;
    o[(size_t)co * T * Ci + (size_t)t * Ci + ci] = (f16)w[i];
}

// Combined QKV projection (fold mha in_proj into q/k/v projections): [48 x 192]
__global__ void k_prep_qkvw(const float* __restrict__ in_w, const float* __restrict__ in_b,
                            const float* __restrict__ qw, const float* __restrict__ qb,
                            const float* __restrict__ kw, const float* __restrict__ kb,
                            const float* __restrict__ vw, const float* __restrict__ vb,
                            f16* __restrict__ Wc, float* __restrict__ bc) {
    int i = blockIdx.x * blockDim.x + threadIdx.x;
    if (i >= 48 * kNS) return;
    int eo = i / kNS, c = i % kNS;
    int sel = eo / 16;
    const float* W  = (sel == 0) ? qw : (sel == 1) ? kw : vw;
    const float* Bv = (sel == 0) ? qb : (sel == 1) ? kb : vb;
    float acc = 0.f;
    for (int f = 0; f < 16; f++) acc += in_w[eo * 16 + f] * W[f * kNS + c];
    Wc[i] = (f16)acc;
    if (c == 0) {
        float bb = in_b[eo];
        for (int f = 0; f < 16; f++) bb += in_w[eo * 16 + f] * Bv[f];
        bc[eo] = bb;
    }
}

// Combined output projection (op_w then out_w): [192 x 16] f32
__global__ void k_prep_outw(const float* __restrict__ out_w, const float* __restrict__ out_b,
                            const float* __restrict__ op_w, const float* __restrict__ op_b,
                            float* __restrict__ cw, float* __restrict__ cb) {
    int i = blockIdx.x * blockDim.x + threadIdx.x;
    if (i >= kNS * 16) return;
    int c = i / 16, e = i % 16;
    float acc = 0.f;
    for (int f = 0; f < 16; f++) acc += out_w[c * 16 + f] * op_w[f * 16 + e];
    cw[i] = acc;
    if (e == 0) {
        float bb = out_b[c];
        for (int f = 0; f < 16; f++) bb += out_w[c * 16 + f] * op_b[f];
        cb[c] = bb;
    }
}

// ---------------------------------------------------------------------------
// Cross-attention collapses exactly: all keys identical -> softmax uniform ->
// output = projected V vector per batch. ca_vec[b][0..191].
// ---------------------------------------------------------------------------
__global__ void k_cross(const float* __restrict__ text, const float* __restrict__ cond,
                        const float* __restrict__ comb_w, const float* __restrict__ comb_b,
                        const float* __restrict__ cng, const float* __restrict__ cnb,
                        const float* __restrict__ vw, const float* __restrict__ vb,
                        const float* __restrict__ in_w, const float* __restrict__ in_b,
                        const float* __restrict__ op_w, const float* __restrict__ op_b,
                        const float* __restrict__ out_w, const float* __restrict__ out_b,
                        float* __restrict__ ca_vec) {
    __shared__ float ce[kNS], v1[16], v2[16], hh[16];
    int tid = threadIdx.x;  // 192
    for (int b = 0; b < kB; b++) {
        float acc = comb_b[tid];
        const float* wr = comb_w + (size_t)tid * (kTD + 3);
        const float* te = text + b * kTD;
        for (int j = 0; j < kTD; j++) acc += wr[j] * te[j];
        for (int j = 0; j < 3; j++)  acc += wr[kTD + j] * cond[b * 3 + j];
        ce[tid] = acc;
        __syncthreads();
        float sum = 0.f, sq = 0.f;
        for (int c = 0; c < kNS; c++) { sum += ce[c]; sq += ce[c] * ce[c]; }
        float mean = sum / (float)kNS;
        float var  = sq / (float)kNS - mean * mean;
        float cen  = (ce[tid] - mean) * rsqrtf(var + 1e-5f) * cng[tid] + cnb[tid];
        __syncthreads();
        ce[tid] = cen;
        __syncthreads();
        if (tid < 16) {
            float a = vb[tid];
            for (int c = 0; c < kNS; c++) a += vw[tid * kNS + c] * ce[c];
            v1[tid] = a;
        }
        __syncthreads();
        if (tid < 16) {
            float a = in_b[32 + tid];
            for (int f = 0; f < 16; f++) a += in_w[(32 + tid) * 16 + f] * v1[f];
            v2[tid] = a;
        }
        __syncthreads();
        if (tid < 16) {
            float a = op_b[tid];
            for (int f = 0; f < 16; f++) a += op_w[tid * 16 + f] * v2[f];
            hh[tid] = a;
        }
        __syncthreads();
        float a = out_b[tid];
        for (int e = 0; e < 16; e++) a += out_w[tid * 16 + e] * hh[e];
        ca_vec[b * kNS + tid] = a;
        __syncthreads();
    }
}

// ---------------------------------------------------------------------------
// Time embedding MLP -> tvec [B,128]
// ---------------------------------------------------------------------------
__global__ void k_time(const int* __restrict__ t,
                       const float* __restrict__ w1, const float* __restrict__ b1,
                       const float* __restrict__ w2, const float* __restrict__ b2,
                       float* __restrict__ tvec) {
    __shared__ float emb[64], h1[128];
    int tid = threadIdx.x;  // 128
    for (int b = 0; b < kB; b++) {
        if (tid < 32) {
            float fr = __expf((float)tid * (-10.f / 31.f));
            float e  = (float)(t[b] + 1) * fr;
            emb[tid]      = sinf(e);
            emb[tid + 32] = cosf(e);
        }
        __syncthreads();
        float a = b1[tid];
        for (int j = 0; j < 64; j++) a += w1[tid * 64 + j] * emb[j];
        h1[tid] = mishf(a);
        __syncthreads();
        float a2 = b2[tid];
        for (int j = 0; j < 128; j++) a2 += w2[tid * 128 + j] * h1[j];
        tvec[b * 128 + tid] = a2;
        __syncthreads();
    }
}

// xp = x + tvec broadcast, cast to f16 (feeds pre-conv and res-conv GEMMs)
__global__ void k_addtime(const float* __restrict__ x, const float* __restrict__ tvec,
                          f16* __restrict__ xph) {
    size_t i = (size_t)blockIdx.x * blockDim.x + threadIdx.x;
    size_t n = (size_t)kB * kNI * kL;
    if (i >= n) return;
    int b  = (int)(i / ((size_t)kNI * kL));
    int ci = (int)((i / kL) % kNI);
    xph[i] = (f16)(x[i] + tvec[b * kNI + ci]);
}

// ---------------------------------------------------------------------------
// Implicit-GEMM conv1d (SAME, stride 1) via v_wmma_f32_16x16x32_f16.
// Fully compile-time shapes/strides:
//  - SXC==1 ([B,L,C] input): B fragment is one 32-byte vector load (b128 x2)
//  - SXC!=1 ([B,C,L] input): immediate-offset clause of u16 loads, no addr math
//  - 2-wide N blocking: A fragment reused across two WMMAs per K-chunk
//  - T==1 instantiations have no boundary branch at all (EXEC stays full)
// W is f16 [Co][T][Ci].
// ---------------------------------------------------------------------------
template <int CI, int T, int PAD,
          long SXB, long SXC, long SXL,
          long SYB, long SYC, long SYL>
__global__ void k_convgemm(const f16* __restrict__ W, const float* __restrict__ bias,
                           const f16* __restrict__ X, float* __restrict__ Y) {
    const int lane = threadIdx.x;  // 32
    const int h16  = lane >> 4;
    const int ln   = lane & 15;
    const int tn = blockIdx.x, tm = blockIdx.y, b = blockIdx.z;

    v8f acc0 = {0.f, 0.f, 0.f, 0.f, 0.f, 0.f, 0.f, 0.f};
    v8f acc1 = {0.f, 0.f, 0.f, 0.f, 0.f, 0.f, 0.f, 0.f};

    const int arow = tm * 16 + ln;                 // A row (output channel)
    const f16* Wrow = W + (size_t)arow * T * CI;
    const f16* Xb   = X + (size_t)b * SXB;
    const int n0 = tn * 32 + ln;                   // output columns of the two tiles
    const int n1 = n0 + 16;

    __builtin_prefetch(Wrow, 0, 1);                // global_prefetch_b8
    __builtin_prefetch(Xb + (size_t)n0 * SXL, 0, 1);

#pragma unroll
    for (int t = 0; t < T; t++) {
        const int l0 = n0 + t - PAD;               // shifted input columns
        const int l1 = n1 + t - PAD;
        const bool v0 = (T == 1) || ((l0 >= 0) && (l0 < kL));
        const bool v1 = (T == 1) || ((l1 >= 0) && (l1 < kL));
        const f16* Wt = Wrow + (size_t)t * CI;
#pragma unroll
        for (int kc = 0; kc < CI; kc += 32) {
            v16h a;
            // A fragment (contiguous k, matched by B): lane holds row, K cols {16h..16h+15}
#pragma unroll
            for (int j = 0; j < 8; j++) {
                a[j]     = Wt[kc + 16 * h16 + j];
                a[8 + j] = Wt[kc + 16 * h16 + 8 + j];
            }
            // B fragment (same contiguous k as A): lane holds col, K cols {16h..16h+15}
            v16h b0, b1;
            if constexpr (SXC == 1) {
                b0 = v0 ? *reinterpret_cast<const v16h*>(
                              Xb + (size_t)(kc + 16 * h16) + (size_t)l0 * SXL)
                        : h16zero();
                b1 = v1 ? *reinterpret_cast<const v16h*>(
                              Xb + (size_t)(kc + 16 * h16) + (size_t)l1 * SXL)
                        : h16zero();
            } else {
                if (v0) {
                    const f16* p = Xb + (size_t)(kc + 16 * h16) * SXC + (size_t)l0 * SXL;
#pragma unroll
                    for (int j = 0; j < 16; j++) b0[j] = p[(size_t)j * SXC];
                } else b0 = h16zero();
                if (v1) {
                    const f16* p = Xb + (size_t)(kc + 16 * h16) * SXC + (size_t)l1 * SXL;
#pragma unroll
                    for (int j = 0; j < 16; j++) b1[j] = p[(size_t)j * SXC];
                } else b1 = h16zero();
            }
            acc0 = __builtin_amdgcn_wmma_f32_16x16x32_f16(false, a, false, b0,
                                                          (short)0, acc0, false, false);
            acc1 = __builtin_amdgcn_wmma_f32_16x16x32_f16(false, a, false, b1,
                                                          (short)0, acc1, false, false);
        }
    }

    // C/D layout: VGPR j -> row (tm*16 + j + 8*h16), col = lane&15
    const int corow = tm * 16 + 8 * h16;
    float bv[8];
#pragma unroll
    for (int j = 0; j < 8; j++) bv[j] = bias[corow + j];
    float* Yb = Y + (size_t)b * SYB;

    if constexpr (SYC == 1) {
        // channels contiguous in memory -> two float4 stores per tile
        float* p0 = Yb + (size_t)corow + (size_t)n0 * SYL;
        float* p1 = Yb + (size_t)corow + (size_t)n1 * SYL;
        *reinterpret_cast<float4*>(p0) =
            make_float4(acc0[0] + bv[0], acc0[1] + bv[1], acc0[2] + bv[2], acc0[3] + bv[3]);
        *reinterpret_cast<float4*>(p0 + 4) =
            make_float4(acc0[4] + bv[4], acc0[5] + bv[5], acc0[6] + bv[6], acc0[7] + bv[7]);
        *reinterpret_cast<float4*>(p1) =
            make_float4(acc1[0] + bv[0], acc1[1] + bv[1], acc1[2] + bv[2], acc1[3] + bv[3]);
        *reinterpret_cast<float4*>(p1 + 4) =
            make_float4(acc1[4] + bv[4], acc1[5] + bv[5], acc1[6] + bv[6], acc1[7] + bv[7]);
    } else {
#pragma unroll
        for (int j = 0; j < 8; j++) {
            Yb[(size_t)(corow + j) * SYC + (size_t)n0 * SYL] = acc0[j] + bv[j];
            Yb[(size_t)(corow + j) * SYC + (size_t)n1 * SYL] = acc1[j] + bv[j];
        }
    }
}

// ---------------------------------------------------------------------------
// GroupNorm(1) stats: deterministic two-stage reduction (no float atomics)
// ---------------------------------------------------------------------------
__global__ void k_stats(const float* __restrict__ x, float* __restrict__ part, int nper) {
    __shared__ float sh[256], sh2[256];
    int b = blockIdx.y;
    const float* p = x + (size_t)b * nper;
    float s = 0.f, q = 0.f;
    for (int i = blockIdx.x * blockDim.x + threadIdx.x; i < nper;
         i += gridDim.x * blockDim.x) {
        float v = p[i];
        s += v; q += v * v;
    }
    sh[threadIdx.x] = s; sh2[threadIdx.x] = q;
    __syncthreads();
    for (int o = 128; o > 0; o >>= 1) {
        if ((int)threadIdx.x < o) {
            sh[threadIdx.x]  += sh[threadIdx.x + o];
            sh2[threadIdx.x] += sh2[threadIdx.x + o];
        }
        __syncthreads();
    }
    if (threadIdx.x == 0) {
        part[((size_t)b * gridDim.x + blockIdx.x) * 2 + 0] = sh[0];
        part[((size_t)b * gridDim.x + blockIdx.x) * 2 + 1] = sh2[0];
    }
}

__global__ void k_stats2(const float* __restrict__ part, float* __restrict__ stats, int nblk) {
    int tid = threadIdx.x;
    if (tid < kB) {
        float s = 0.f, q = 0.f;
        for (int i = 0; i < nblk; i++) {
            s += part[((size_t)tid * nblk + i) * 2 + 0];
            q += part[((size_t)tid * nblk + i) * 2 + 1];
        }
        stats[2 * tid] = s; stats[2 * tid + 1] = q;
    }
}

// GN(1 group) + mish -> f16
__global__ void k_gn_mish_h(const float* __restrict__ x, f16* __restrict__ y,
                            const float* __restrict__ stats, const float* __restrict__ g,
                            const float* __restrict__ bt, int C, float invN) {
    size_t i = (size_t)blockIdx.x * blockDim.x + threadIdx.x;
    size_t n = (size_t)kB * C * kL;
    if (i >= n) return;
    int b = (int)(i / ((size_t)C * kL));
    int c = (int)((i / kL) % C);
    float mean = stats[2 * b] * invN;
    float var  = stats[2 * b + 1] * invN - mean * mean;
    float v = (x[i] - mean) * rsqrtf(var + 1e-5f) * g[c] + bt[c];
    y[i] = (f16)mishf(v);
}

// LayerNorm over channel dim of s [B,L,192] (+ optional per-batch add) -> f16
__global__ void k_ln_h(const float* __restrict__ s, const float* __restrict__ addv,
                       const float* __restrict__ g, const float* __restrict__ bt,
                       f16* __restrict__ y) {
    int idx = blockIdx.x * blockDim.x + threadIdx.x;  // B*L
    if (idx >= kB * kL) return;
    int b = idx / kL;
    const float* base = s + (size_t)idx * kNS;
    const float* av = addv ? (addv + b * kNS) : nullptr;
    float sum = 0.f, sq = 0.f;
    for (int c = 0; c < kNS; c++) {
        float v = base[c] + (av ? av[c] : 0.f);
        sum += v; sq += v * v;
    }
    float mean = sum / (float)kNS;
    float var  = sq / (float)kNS - mean * mean;
    float rstd = rsqrtf(var + 1e-5f);
    for (int c = 0; c < kNS; c++) {
        float v = base[c] + (av ? av[c] : 0.f);
        y[(size_t)idx * kNS + c] = (f16)((v - mean) * rstd * g[c] + bt[c]);
    }
}

// ---------------------------------------------------------------------------
// Flash self-attention, one wave per (b, head, 16-query tile).
// S^T = K_tile(16x4) x Q^T(4x16) via v_wmma_f32_16x16x4_f32 (full f32).
// C-layout puts query in the lane index -> online softmax is per-lane work
// plus one shfl_xor(16) to merge key halves.
// qkv layout: [B, L, 48] = [q(16) | k(16) | v(16)], head h at dims 4h..4h+3.
// ---------------------------------------------------------------------------
__global__ void k_attn(const float* __restrict__ qkv, float* __restrict__ hbuf) {
    int lane = threadIdx.x;          // 32
    int tile = blockIdx.x;           // L/16 query tiles
    int head = blockIdx.y;
    int b    = blockIdx.z;
    int h16  = lane >> 4;
    int ln   = lane & 15;
    const float scale = 0.5f;        // 1/sqrt(hd=4)

    size_t rowQ = ((size_t)b * kL + (size_t)tile * 16 + ln) * 48;
    v2f bq;
    bq.x = qkv[rowQ + head * 4 + 2 * h16 + 0] * scale;   // B: Q^T[dim][query]
    bq.y = qkv[rowQ + head * 4 + 2 * h16 + 1] * scale;

    float m_run = -3e38f, s_run = 0.f;
    float o0 = 0.f, o1 = 0.f, o2 = 0.f, o3 = 0.f;

    for (int kt = 0; kt < kL / 16; kt++) {
        size_t rowK = ((size_t)b * kL + (size_t)kt * 16 + ln) * 48;
        v2f ak;
        ak.x = qkv[rowK + 16 + head * 4 + 2 * h16 + 0];  // A: K_tile[key][dim]
        ak.y = qkv[rowK + 16 + head * 4 + 2 * h16 + 1];
        v8f sp = {0.f, 0.f, 0.f, 0.f, 0.f, 0.f, 0.f, 0.f};
        sp = __builtin_amdgcn_wmma_f32_16x16x4_f32(false, ak, false, bq,
                                                   (short)0, sp, false, false);
        // lane owns query ln; sp[j] = score(key = kt*16 + 8*h16 + j, query ln)
        float mt = sp[0];
#pragma unroll
        for (int j = 1; j < 8; j++) mt = fmaxf(mt, sp[j]);
        mt = fmaxf(mt, __shfl_xor(mt, 16, 32));
        float m_new = fmaxf(m_run, mt);
        float corr  = __expf(m_run - m_new);
        float p[8];
        float ps = 0.f;
#pragma unroll
        for (int j = 0; j < 8; j++) { p[j] = __expf(sp[j] - m_new); ps += p[j]; }
        ps += __shfl_xor(ps, 16, 32);
        s_run = s_run * corr + ps;
        o0 *= corr; o1 *= corr; o2 *= corr; o3 *= corr;
#pragma unroll
        for (int j = 0; j < 8; j++) {
            const float4 vv = *reinterpret_cast<const float4*>(
                &qkv[((size_t)b * kL + (size_t)kt * 16 + 8 * h16 + j) * 48 + 32 + head * 4]);
            o0 += p[j] * vv.x; o1 += p[j] * vv.y; o2 += p[j] * vv.z; o3 += p[j] * vv.w;
        }
        m_run = m_new;
    }
    o0 += __shfl_xor(o0, 16, 32);
    o1 += __shfl_xor(o1, 16, 32);
    o2 += __shfl_xor(o2, 16, 32);
    o3 += __shfl_xor(o3, 16, 32);
    if (lane < 16) {
        float inv = 1.f / s_run;
        size_t oi = ((size_t)b * kL + (size_t)tile * 16 + ln) * kE + head * 4;
        hbuf[oi + 0] = o0 * inv; hbuf[oi + 1] = o1 * inv;
        hbuf[oi + 2] = o2 * inv; hbuf[oi + 3] = o3 * inv;
    }
}

// s += hbuf @ comb_out^T + comb_out_b  (fused out-proj + residual add)
__global__ void k_saout(float* __restrict__ s, const float* __restrict__ hbuf,
                        const float* __restrict__ cw, const float* __restrict__ cb) {
    size_t i = (size_t)blockIdx.x * blockDim.x + threadIdx.x;
    size_t n = (size_t)kB * kL * kNS;
    if (i >= n) return;
    size_t bl = i / kNS;
    int c = (int)(i % kNS);
    const float* h = hbuf + bl * kE;
    float acc = cb[c];
#pragma unroll
    for (int e = 0; e < 16; e++) acc += cw[c * 16 + e] * h[e];
    s[i] += acc;
}

// out = mish(GN2(ppost)) + res_conv  -> d_out (f32)
__global__ void k_final(const float* __restrict__ p, const float* __restrict__ r,
                        const float* __restrict__ stats, const float* __restrict__ g,
                        const float* __restrict__ bt, float* __restrict__ out) {
    size_t i = (size_t)blockIdx.x * blockDim.x + threadIdx.x;
    size_t n = (size_t)kB * kNO * kL;
    if (i >= n) return;
    int b = (int)(i / ((size_t)kNO * kL));
    int c = (int)((i / kL) % kNO);
    const float invN = 1.f / ((float)kNO * (float)kL);
    float mean = stats[2 * b] * invN;
    float var  = stats[2 * b + 1] * invN - mean * mean;
    float v = (p[i] - mean) * rsqrtf(var + 1e-5f) * g[c] + bt[c];
    out[i] = mishf(v) + r[i];
}

// ---------------------------------------------------------------------------
extern "C" void kernel_launch(void* const* d_in, const int* in_sizes, int n_in,
                              void* d_out, int out_size, void* d_ws, size_t ws_size,
                              hipStream_t stream) {
    (void)in_sizes; (void)n_in; (void)out_size; (void)ws_size;
    // Inputs per setup_inputs() order
    const float* x_in    = (const float*)d_in[0];
    const int*   t_in    = (const int*)d_in[1];
    const float* text    = (const float*)d_in[2];
    const float* cond    = (const float*)d_in[3];
    const float* te1w = (const float*)d_in[4];  const float* te1b = (const float*)d_in[5];
    const float* te2w = (const float*)d_in[6];  const float* te2b = (const float*)d_in[7];
    const float* prew = (const float*)d_in[8];  const float* preb = (const float*)d_in[9];
    const float* scw  = (const float*)d_in[10]; const float* scb  = (const float*)d_in[11];
    const float* postw= (const float*)d_in[12]; const float* postb= (const float*)d_in[13];
    const float* resw = (const float*)d_in[14]; const float* resb = (const float*)d_in[15];
    const float* gn1g = (const float*)d_in[16]; const float* gn1b = (const float*)d_in[17];
    const float* gn2g = (const float*)d_in[18]; const float* gn2b = (const float*)d_in[19];
    const float* salng= (const float*)d_in[20]; const float* salnb= (const float*)d_in[21];
    const float* saqw = (const float*)d_in[22]; const float* saqb = (const float*)d_in[23];
    const float* sakw = (const float*)d_in[24]; const float* sakb = (const float*)d_in[25];
    const float* savw = (const float*)d_in[26]; const float* savb = (const float*)d_in[27];
    const float* saow = (const float*)d_in[28]; const float* saob = (const float*)d_in[29];
    const float* saiw = (const float*)d_in[30]; const float* saib = (const float*)d_in[31];
    const float* sapw = (const float*)d_in[32]; const float* sapb = (const float*)d_in[33];
    // 34,35: ca_ln_{g,b}; 36,37: ca_q_{w,b} — unused (cross-attn q irrelevant)
    const float* cavw = (const float*)d_in[40]; const float* cavb = (const float*)d_in[41];
    const float* caow = (const float*)d_in[42]; const float* caob = (const float*)d_in[43];
    const float* caiw = (const float*)d_in[44]; const float* caib = (const float*)d_in[45];
    const float* capw = (const float*)d_in[46]; const float* capb = (const float*)d_in[47];
    const float* cacw = (const float*)d_in[48]; const float* cacb = (const float*)d_in[49];
    const float* cang = (const float*)d_in[50]; const float* canb = (const float*)d_in[51];
    const float* ang  = (const float*)d_in[52]; const float* anb  = (const float*)d_in[53];
    float* out = (float*)d_out;

    // --- ws bump allocator (deterministic layout every call) ---
    char* wsb = (char*)d_ws;
    size_t off = 0;
    auto alloc = [&](size_t bytes) -> char* {
        char* p = wsb + off;
        off += (bytes + 255) & ~(size_t)255;
        return p;
    };
    f16*   w16_pre  = (f16*)alloc((size_t)kNS * 5 * kNI * 2);
    f16*   w16_sc   = (f16*)alloc((size_t)kNS * 1 * kNS * 2);
    f16*   w16_post = (f16*)alloc((size_t)kNO * 5 * kNS * 2);
    f16*   w16_res  = (f16*)alloc((size_t)kNO * 1 * kNI * 2);
    f16*   w16_qkv  = (f16*)alloc((size_t)48 * kNS * 2);
    float* b_qkv    = (float*)alloc(48 * 4);
    float* comb_ow  = (float*)alloc((size_t)kNS * 16 * 4);
    float* comb_ob  = (float*)alloc(kNS * 4);
    float* ca_vec   = (float*)alloc(kB * kNS * 4);
    float* tvec     = (float*)alloc(kB * kNI * 4);
    float* stats    = (float*)alloc(16 * 4);          // [gn1: 8][gn2: 8]
    float* part     = (float*)alloc((size_t)kB * 64 * 2 * 4);
    f16*   xph      = (f16*)alloc((size_t)kB * kNI * kL * 2);
    float* s1       = (float*)alloc((size_t)kB * kNS * kL * 4);
    f16*   s1h      = (f16*)alloc((size_t)kB * kNS * kL * 2);
    float* sbuf     = (float*)alloc((size_t)kB * kL * kNS * 4);
    f16*   xnh      = (f16*)alloc((size_t)kB * kL * kNS * 2);
    float* qkvbuf   = (float*)alloc((size_t)kB * kL * 48 * 4);
    float* hbuf     = (float*)alloc((size_t)kB * kL * kE * 4);
    f16*   snh      = (f16*)alloc((size_t)kB * kL * kNS * 2);
    float* ppost    = (float*)alloc((size_t)kB * kNO * kL * 4);
    float* rbuf     = (float*)alloc((size_t)kB * kNO * kL * 4);

    auto blk = [](size_t n) { return dim3((unsigned)((n + 255) / 256)); };

    // --- prep (params only) ---
    k_prep_convw<<<blk((size_t)kNS * kNI * 5), 256, 0, stream>>>(prew, w16_pre, kNS, kNI, 5);
    k_prep_convw<<<blk((size_t)kNS * kNS * 1), 256, 0, stream>>>(scw, w16_sc, kNS, kNS, 1);
    k_prep_convw<<<blk((size_t)kNO * kNS * 5), 256, 0, stream>>>(postw, w16_post, kNO, kNS, 5);
    k_prep_convw<<<blk((size_t)kNO * kNI * 1), 256, 0, stream>>>(resw, w16_res, kNO, kNI, 1);
    k_prep_qkvw<<<blk(48 * kNS), 256, 0, stream>>>(saiw, saib, saqw, saqb, sakw, sakb,
                                                   savw, savb, w16_qkv, b_qkv);
    k_prep_outw<<<blk(kNS * 16), 256, 0, stream>>>(saow, saob, sapw, sapb, comb_ow, comb_ob);
    k_cross<<<1, kNS, 0, stream>>>(text, cond, cacw, cacb, cang, canb,
                                   cavw, cavb, caiw, caib, capw, capb, caow, caob, ca_vec);
    k_time<<<1, 128, 0, stream>>>(t_in, te1w, te1b, te2w, te2b, tvec);

    // --- main path ---
    k_addtime<<<blk((size_t)kB * kNI * kL), 256, 0, stream>>>(x_in, tvec, xph);

    // pre conv: xph [B,128,L] -> s1 [B,192,L]
    k_convgemm<128, 5, 2,
               (long)kNI * kL, (long)kL, 1L,
               (long)kNS * kL, (long)kL, 1L>
        <<<dim3(kL / 32, kNS / 16, kB), 32, 0, stream>>>(w16_pre, preb, xph, s1);

    k_stats<<<dim3(64, kB), 256, 0, stream>>>(s1, part, kNS * kL);
    k_stats2<<<1, 64, 0, stream>>>(part, stats, 64);
    k_gn_mish_h<<<blk((size_t)kB * kNS * kL), 256, 0, stream>>>(
        s1, s1h, stats, gn1g, gn1b, kNS, 1.f / ((float)kNS * kL));

    // sc conv 1x1: s1h [B,192,L] -> sbuf [B,L,192]
    k_convgemm<192, 1, 0,
               (long)kNS * kL, (long)kL, 1L,
               (long)kL * kNS, 1L, (long)kNS>
        <<<dim3(kL / 32, kNS / 16, kB), 32, 0, stream>>>(w16_sc, scb, s1h, sbuf);

    // self-attn: LN -> combined QKV GEMM -> flash attn -> fused out-proj add
    k_ln_h<<<blk(kB * kL), 256, 0, stream>>>(sbuf, nullptr, salng, salnb, xnh);
    k_convgemm<192, 1, 0,
               (long)kL * kNS, 1L, (long)kNS,
               (long)kL * 48, 1L, 48L>
        <<<dim3(kL / 32, 48 / 16, kB), 32, 0, stream>>>(w16_qkv, b_qkv, xnh, qkvbuf);
    k_attn<<<dim3(kL / 16, 4, kB), 32, 0, stream>>>(qkvbuf, hbuf);
    k_saout<<<blk((size_t)kB * kL * kNS), 256, 0, stream>>>(sbuf, hbuf, comb_ow, comb_ob);

    // cross-attn add (per-batch vector) fused into final LN -> snh f16
    k_ln_h<<<blk(kB * kL), 256, 0, stream>>>(sbuf, ca_vec, ang, anb, snh);

    // post conv: snh [B,L,192] -> ppost [B,256,L]
    k_convgemm<192, 5, 2,
               (long)kL * kNS, 1L, (long)kNS,
               (long)kNO * kL, (long)kL, 1L>
        <<<dim3(kL / 32, kNO / 16, kB), 32, 0, stream>>>(w16_post, postb, snh, ppost);

    k_stats<<<dim3(64, kB), 256, 0, stream>>>(ppost, part, kNO * kL);
    k_stats2<<<1, 64, 0, stream>>>(part, stats + 8, 64);

    // residual 1x1 conv: xph [B,128,L] -> rbuf [B,256,L]
    k_convgemm<128, 1, 0,
               (long)kNI * kL, (long)kL, 1L,
               (long)kNO * kL, (long)kL, 1L>
        <<<dim3(kL / 32, kNO / 16, kB), 32, 0, stream>>>(w16_res, resb, xph, rbuf);

    // out = mish(GN2(ppost)) + rbuf
    k_final<<<blk((size_t)kB * kNO * kL), 256, 0, stream>>>(
        ppost, rbuf, stats + 8, gn2g, gn2b, out);
}
